// GenericCrossModalFusionKernel_59622736003290
// MI455X (gfx1250) — matrix operand order repaired and checked by reading, not compiled, not run on hardware
//
#include <hip/hip_runtime.h>
#include <hip/hip_bf16.h>

// ---------------------------------------------------------------------------
// Cross-modal fusion for MI455X (gfx1250, wave32, WMMA).
//
// ~275 GFLOP of GEMM vs ~600 MB traffic -> compute bound at 23.3 TB/s, so all
// GEMMs run through v_wmma_f32_16x16x32_bf16. fp32 softmax/norms/epilogues.
// Gate GEMMs: the broadcast-average half of the concat collapses to a
// per-(b,h) bias GEMV, halving gate GEMM K from 4096 to 2048.
//
// GEMM: block tile 256x128x32, 8 waves (4 along M x 2 along N), wave tile
// 64x64 -> 16 WMMAs per 16 ds_load_b128 per K-step. Double-buffered LDS with
// one barrier per K-step so global loads for tile i+1 overlap WMMAs of tile i.
// ---------------------------------------------------------------------------

typedef __attribute__((ext_vector_type(16))) __bf16 bf16x16;
typedef __attribute__((ext_vector_type(8)))  float  f32x8;
typedef __attribute__((ext_vector_type(4)))  float  f32x4;

union ABFrag { f32x4 f4[2]; bf16x16 v; };

static constexpr int BATCH = 4;
static constexpr int VSL   = 1024;   // vision seq len
static constexpr int LSL   = 2048;   // language seq len
static constexpr int HD    = 2048;   // hidden
static constexpr int ML    = 2048;   // max_len (padded vision rows)

#define TILE_M 256
#define TILE_N 128
#define TILE_K 32
#define LDSS   40   // bf16 elems per LDS tile row (80B, 16B-aligned frag reads)

// ---------------------------------------------------------------------------
// NT GEMM:  C[m,n] = sum_k A[m,k] * B[n,k]   (A,B bf16 row-major, f32 accum)
// EPI=0: store f32 C.   EPI=1: out = feat + fused * sigmoid(C + bias[n]).
// ---------------------------------------------------------------------------
template<int EPI>
__global__ __launch_bounds__(256)
void gemm_nt_bf16(const __bf16* __restrict__ A, long aBatch, int lda,
                  const __bf16* __restrict__ B, long bBatch, int ldb,
                  float* __restrict__ C, long cBatch, int ldc,
                  const float* __restrict__ bias,
                  const float* __restrict__ fused, long fBatch,
                  const float* __restrict__ feat,  long featBatch,
                  float* __restrict__ outp, long oBatch,
                  int K)
{
    __shared__ __bf16 As[2][TILE_M * LDSS];   // 2 x 20 KB
    __shared__ __bf16 Bs[2][TILE_N * LDSS];   // 2 x 10 KB

    const int b     = blockIdx.z;
    const int tid   = threadIdx.x;
    const int lane  = tid & 31;
    const int wave  = tid >> 5;
    const int wm    = wave & 3;      // 4 waves along M -> 64 rows each
    const int wn    = wave >> 2;     // 2 waves along N -> 64 cols each
    const int l15   = lane & 15;
    const int lhalf = lane >> 4;

    const int tileM = blockIdx.y * TILE_M;
    const int tileN = blockIdx.x * TILE_N;

    const __bf16* Ag = A + (long)b * aBatch + (long)tileM * lda;
    const __bf16* Bg = B + (long)b * bBatch + (long)tileN * ldb;

    // Tile staging: A 256x32 (1024 float4), B 128x32 (512 float4); per thread
    // 4 + 2 float4. Four consecutive lanes cover one 64B row segment.
    const int lr = tid >> 2;           // base row 0..63
    const int lc = (tid & 3) * 8;      // bf16 col 0/8/16/24
    f32x4 sA[4], sB[2];

    auto load_global = [&](int k0) {
#pragma unroll
        for (int p = 0; p < 4; ++p)
            sA[p] = *(const f32x4*)(Ag + (long)(lr + p * 64) * lda + k0 + lc);
#pragma unroll
        for (int p = 0; p < 2; ++p)
            sB[p] = *(const f32x4*)(Bg + (long)(lr + p * 64) * ldb + k0 + lc);
    };
    auto store_lds = [&](int buf) {
#pragma unroll
        for (int p = 0; p < 4; ++p)
            *(f32x4*)(&As[buf][(lr + p * 64) * LDSS + lc]) = sA[p];
#pragma unroll
        for (int p = 0; p < 2; ++p)
            *(f32x4*)(&Bs[buf][(lr + p * 64) * LDSS + lc]) = sB[p];
    };

    f32x8 acc[4][4] = {};

    const int nIter = K / TILE_K;
    load_global(0);
    store_lds(0);
    if (nIter > 1) load_global(TILE_K);
    __syncthreads();

    for (int i = 0; i < nIter; ++i) {
        const int buf = i & 1;

        // near-cache prefetch for tile i+2 (speculative; OOB is dropped)
        __builtin_prefetch(Ag + (long)lr * lda + (i + 2) * TILE_K, 0, 3);
        __builtin_prefetch(Bg + (long)lr * ldb + (i + 2) * TILE_K, 0, 3);

        // B frags: lane n<16 -> col n, K0..15; lane n+16 -> K16..31 (contig 32B)
        ABFrag bfr[4];
#pragma unroll
        for (int fn = 0; fn < 4; ++fn) {
            const __bf16* p = &Bs[buf][(wn * 64 + fn * 16 + l15) * LDSS + lhalf * 16];
            bfr[fn].f4[0] = *(const f32x4*)(p);
            bfr[fn].f4[1] = *(const f32x4*)(p + 8);
        }
        // A frags: lane L<16 -> row L, K0..15; lane L+16 -> K16..31 (contig 32B)
#pragma unroll
        for (int fm = 0; fm < 4; ++fm) {
            ABFrag af;
            const __bf16* p = &As[buf][(wm * 64 + fm * 16 + l15) * LDSS + lhalf * 16];
            af.f4[0] = *(const f32x4*)(p);
            af.f4[1] = *(const f32x4*)(p + 8);
#pragma unroll
            for (int fn = 0; fn < 4; ++fn)
                acc[fm][fn] = __builtin_amdgcn_wmma_f32_16x16x32_bf16(
                    false, af.v, false, bfr[fn].v,
                    (short)0, acc[fm][fn], false, false);
        }

        if (i + 1 < nIter) {
            // buf^1 was last read in iteration i-1; its trailing barrier makes
            // these stores safe. Then issue global loads for tile i+2.
            store_lds(buf ^ 1);
            if (i + 2 < nIter) load_global((i + 2) * TILE_K);
        }
        __syncthreads();
    }

    // C layout: VGPR r, lanes 0-15 -> M=r, N=lane; lanes 16-31 -> M=r+8, N=lane-16
    if (EPI == 0) {
        float* Cp = C + (long)b * cBatch;
#pragma unroll
        for (int fm = 0; fm < 4; ++fm)
#pragma unroll
            for (int fn = 0; fn < 4; ++fn)
#pragma unroll
                for (int r = 0; r < 8; ++r) {
                    int m = tileM + wm * 64 + fm * 16 + r + 8 * lhalf;
                    int n = tileN + wn * 64 + fn * 16 + l15;
                    Cp[(long)m * ldc + n] = acc[fm][fn][r];
                }
    } else {
        const float* biasp  = bias  + (long)b * ldc;
        const float* fusedp = fused + (long)b * fBatch;
        const float* featp  = feat  + (long)b * featBatch;
        float*       op     = outp  + (long)b * oBatch;
#pragma unroll
        for (int fm = 0; fm < 4; ++fm)
#pragma unroll
            for (int fn = 0; fn < 4; ++fn)
#pragma unroll
                for (int r = 0; r < 8; ++r) {
                    int m = tileM + wm * 64 + fm * 16 + r + 8 * lhalf;
                    int n = tileN + wn * 64 + fn * 16 + l15;
                    float g = 1.0f / (1.0f + __expf(-(acc[fm][fn][r] + biasp[n])));
                    op[(long)m * ldc + n] =
                        featp[(long)m * ldc + n] + fusedp[(long)m * ldc + n] * g;
                }
    }
}

// ---------------------------------------------------------------------------
// Row L2-normalize -> bf16, plus plain bf16 copy; zero-fill pad rows.
// ---------------------------------------------------------------------------
__global__ __launch_bounds__(256)
void prep_rows(const float* __restrict__ src, long srcBatch, int validRows,
               __bf16* __restrict__ normOut, long normBatch,
               __bf16* __restrict__ plainOut, long plainBatch, int plainRows)
{
    const int b = blockIdx.y, row = blockIdx.x, tid = threadIdx.x;
    __bf16* no = normOut + (long)b * normBatch + (long)row * HD;
    if (row >= validRows) {
        for (int i = tid; i < HD; i += 256) no[i] = (__bf16)0.0f;
        return;
    }
    const float* sp = src + (long)b * srcBatch + (long)row * HD;
    __shared__ float red[256];
    float vals[HD / 256];
    float ss = 0.0f;
#pragma unroll
    for (int j = 0; j < HD / 256; ++j) {
        float x = sp[tid + j * 256];
        vals[j] = x;
        ss += x * x;
    }
    red[tid] = ss; __syncthreads();
    for (int st = 128; st > 0; st >>= 1) {
        if (tid < st) red[tid] += red[tid + st];
        __syncthreads();
    }
    const float inv = 1.0f / fmaxf(sqrtf(red[0]), 1e-12f);
#pragma unroll
    for (int j = 0; j < HD / 256; ++j)
        no[tid + j * 256] = (__bf16)(vals[j] * inv);
    if (plainOut != nullptr && row < plainRows) {
        __bf16* po = plainOut + (long)b * plainBatch + (long)row * HD;
#pragma unroll
        for (int j = 0; j < HD / 256; ++j)
            po[tid + j * 256] = (__bf16)vals[j];
    }
}

// ---------------------------------------------------------------------------
// f32 [rows x cols] -> bf16 transposed [cols x rows], LDS 32x33 tile.
// ---------------------------------------------------------------------------
__global__ __launch_bounds__(256)
void transpose_f32_to_bf16(const float* __restrict__ src, __bf16* __restrict__ dst,
                           int rows, int cols)
{
    __shared__ float tile[32][33];
    const int b = blockIdx.z;
    src += (long)b * rows * cols;
    dst += (long)b * rows * cols;
    const int c0 = blockIdx.x * 32, r0 = blockIdx.y * 32;
    const int tx = threadIdx.x & 31, tg = threadIdx.x >> 5;
#pragma unroll
    for (int i = 0; i < 4; ++i) {
        int r = tg * 4 + i;
        tile[r][tx] = src[(long)(r0 + r) * cols + (c0 + tx)];
    }
    __syncthreads();
#pragma unroll
    for (int i = 0; i < 4; ++i) {
        int r = tg * 4 + i;
        dst[(long)(c0 + r) * rows + (r0 + tx)] = (__bf16)tile[tx][r];
    }
}

// ---------------------------------------------------------------------------
// Column sums (pad rows are zero, so sum of real rows / max_len == padded mean)
// ---------------------------------------------------------------------------
__global__ __launch_bounds__(256)
void colsum_f32(const float* __restrict__ src, float* __restrict__ out,
                int rows, int cols)
{
    const int b = blockIdx.y;
    const int h = blockIdx.x * 256 + threadIdx.x;
    const float* p = src + (long)b * rows * cols;
    float s = 0.0f;
    for (int r = 0; r < rows; ++r) s += p[(long)r * cols + h];
    out[(long)b * cols + h] = s;
}

// ---------------------------------------------------------------------------
// bias[b,h] = (1/ML) * dot(sums[b,:], W[h, 0:HD])   (first half of concat dim)
// 8 waves/block, one (b,h) per wave, shuffle reduction (wave32).
// ---------------------------------------------------------------------------
__global__ __launch_bounds__(256)
void bias_gemv(const float* __restrict__ sums, const float* __restrict__ W,
               int ldw, float scale, float* __restrict__ bias)
{
    const int gid  = blockIdx.x * 8 + (threadIdx.x >> 5);
    const int lane = threadIdx.x & 31;
    const int b = gid >> 11;       // / 2048
    const int h = gid & 2047;
    const float* s = sums + (long)b * HD;
    const float* w = W + (long)h * ldw;
    float acc = 0.0f;
    for (int d = lane; d < HD; d += 32) acc += s[d] * w[d];
    for (int off = 16; off > 0; off >>= 1) acc += __shfl_down(acc, off, 32);
    if (lane == 0) bias[(long)b * HD + h] = acc * scale;
}

// ---------------------------------------------------------------------------
// W[:, koff:koff+HD] -> bf16 (second half of each gate weight).
// ---------------------------------------------------------------------------
__global__ __launch_bounds__(256)
void conv_w_half(const float* __restrict__ W, __bf16* __restrict__ Wb,
                 int ldw, int koff)
{
    const long idx = (long)blockIdx.x * 256 + threadIdx.x;  // over HD*HD
    const int h = (int)(idx >> 11);
    const int d = (int)(idx & 2047);
    Wb[idx] = (__bf16)W[(long)h * ldw + koff + d];
}

// ---------------------------------------------------------------------------
// Row softmax (fp32 math) -> bf16 weights; exponentials cached in registers.
// ---------------------------------------------------------------------------
__global__ __launch_bounds__(256)
void softmax_rows(const float* __restrict__ s, __bf16* __restrict__ w)
{
    __shared__ float red[256];
    const long row = blockIdx.x;
    const int tid = threadIdx.x;
    const float* sp = s + row * LSL;
    __bf16* wp = w + row * LSL;

    float vals[LSL / 256];
    float lmax = -3.4e38f;
#pragma unroll
    for (int j = 0; j < LSL / 256; ++j) {
        vals[j] = sp[tid + j * 256];
        lmax = fmaxf(lmax, vals[j]);
    }
    red[tid] = lmax; __syncthreads();
    for (int st = 128; st > 0; st >>= 1) {
        if (tid < st) red[tid] = fmaxf(red[tid], red[tid + st]);
        __syncthreads();
    }
    const float m = red[0]; __syncthreads();

    float lsum = 0.0f;
#pragma unroll
    for (int j = 0; j < LSL / 256; ++j) {
        vals[j] = __expf(vals[j] - m);
        lsum += vals[j];
    }
    red[tid] = lsum; __syncthreads();
    for (int st = 128; st > 0; st >>= 1) {
        if (tid < st) red[tid] += red[tid + st];
        __syncthreads();
    }
    const float inv = 1.0f / red[0];
#pragma unroll
    for (int j = 0; j < LSL / 256; ++j)
        wp[tid + j * 256] = (__bf16)(vals[j] * inv);
}

// ---------------------------------------------------------------------------
extern "C" void kernel_launch(void* const* d_in, const int* in_sizes, int n_in,
                              void* d_out, int out_size, void* d_ws, size_t ws_size,
                              hipStream_t stream)
{
    (void)in_sizes; (void)n_in; (void)out_size; (void)ws_size;

    const float* vis  = (const float*)d_in[0];   // [4,1024,2048]
    const float* lang = (const float*)d_in[1];   // [4,2048,2048]
    const float* Wvl  = (const float*)d_in[2];   // [2048,4096]
    const float* Wlv  = (const float*)d_in[3];   // [2048,4096]

    float* fused_out = (float*)d_out;                         // [4,2048,2048]
    float* vis_out   = fused_out + (long)BATCH * ML * HD;     // [4,1024,2048]
    float* lang_out  = vis_out   + (long)BATCH * VSL * HD;    // [4,2048,2048]

    char* ws = (char*)d_ws;
    size_t off = 0;
    auto take = [&](size_t bytes) -> void* {
        void* p = ws + off;
        off += (bytes + 255) & ~(size_t)255;
        return p;
    };

    __bf16* vis_nb   = (__bf16*)take((size_t)BATCH * ML  * HD  * 2); // normalized, padded
    __bf16* lang_nb  = (__bf16*)take((size_t)BATCH * LSL * HD  * 2); // normalized
    __bf16* vis_pb   = (__bf16*)take((size_t)BATCH * VSL * HD  * 2); // plain bf16
    __bf16* lang_pb  = (__bf16*)take((size_t)BATCH * LSL * HD  * 2); // plain bf16
    __bf16* langT    = (__bf16*)take((size_t)BATCH * HD  * LSL * 2); // plain, transposed
    __bf16* Wvl2b    = (__bf16*)take((size_t)HD * HD * 2);
    __bf16* Wlv2b    = (__bf16*)take((size_t)HD * HD * 2);
    float*  scores   = (float*) take((size_t)BATCH * ML * LSL * 4);
    __bf16* weights  = (__bf16*)take((size_t)BATCH * ML * LSL * 2);
    float*  vis_sum  = (float*) take((size_t)BATCH * HD * 4);
    float*  lang_sum = (float*) take((size_t)BATCH * HD * 4);
    float*  bias_vl  = (float*) take((size_t)BATCH * HD * 4);
    float*  bias_lv  = (float*) take((size_t)BATCH * HD * 4);

    // 1) normalize + convert (vision pad rows zeroed)
    prep_rows<<<dim3(ML, BATCH), 256, 0, stream>>>(
        vis, (long)VSL * HD, VSL, vis_nb, (long)ML * HD, vis_pb, (long)VSL * HD, VSL);
    prep_rows<<<dim3(LSL, BATCH), 256, 0, stream>>>(
        lang, (long)LSL * HD, LSL, lang_nb, (long)LSL * HD, lang_pb, (long)LSL * HD, LSL);

    // 2) language transposed bf16 for the NN->NT trick in the fused GEMM
    transpose_f32_to_bf16<<<dim3(HD / 32, LSL / 32, BATCH), 256, 0, stream>>>(
        lang, langT, LSL, HD);

    // 3) means (as sums) and collapsed gate biases
    colsum_f32<<<dim3(HD / 256, BATCH), 256, 0, stream>>>(vis,  vis_sum,  VSL, HD);
    colsum_f32<<<dim3(HD / 256, BATCH), 256, 0, stream>>>(lang, lang_sum, LSL, HD);
    bias_gemv<<<dim3(BATCH * HD / 8), 256, 0, stream>>>(vis_sum,  Wvl, 2 * HD, 1.0f / ML, bias_vl);
    bias_gemv<<<dim3(BATCH * HD / 8), 256, 0, stream>>>(lang_sum, Wlv, 2 * HD, 1.0f / ML, bias_lv);

    // 4) gate weight second halves -> bf16
    conv_w_half<<<dim3((HD * HD) / 256), 256, 0, stream>>>(Wvl, Wvl2b, 2 * HD, HD);
    conv_w_half<<<dim3((HD * HD) / 256), 256, 0, stream>>>(Wlv, Wlv2b, 2 * HD, HD);

    // 5) scores = vis_n @ lang_n^T    [4,2048,2048] f32
    gemm_nt_bf16<0><<<dim3(LSL / TILE_N, ML / TILE_M, BATCH), 256, 0, stream>>>(
        vis_nb, (long)ML * HD, HD,
        lang_nb, (long)LSL * HD, HD,
        scores, (long)ML * LSL, LSL,
        nullptr, nullptr, 0, nullptr, 0, nullptr, 0, HD);

    // 6) fp32 softmax -> bf16 weights
    softmax_rows<<<dim3(BATCH * ML), 256, 0, stream>>>(scores, weights);

    // 7) fused = weights @ lang_p  (NT against langT) -> d_out[0]
    gemm_nt_bf16<0><<<dim3(HD / TILE_N, ML / TILE_M, BATCH), 256, 0, stream>>>(
        weights, (long)ML * LSL, LSL,
        langT, (long)HD * LSL, LSL,
        fused_out, (long)ML * HD, HD,
        nullptr, nullptr, 0, nullptr, 0, nullptr, 0, LSL);

    // 8) language_output = lang + fused * sigmoid(lang_p @ Wvl2^T + bias_vl)
    gemm_nt_bf16<1><<<dim3(HD / TILE_N, LSL / TILE_M, BATCH), 256, 0, stream>>>(
        lang_pb, (long)LSL * HD, HD,
        Wvl2b, 0, HD,
        nullptr, 0, HD,
        bias_vl,
        fused_out, (long)ML * HD,
        lang, (long)LSL * HD,
        lang_out, (long)LSL * HD, HD);

    // 9) vision_output = vis + fused[:, :1024] * sigmoid(vis_p @ Wlv2^T + bias_lv)
    gemm_nt_bf16<1><<<dim3(HD / TILE_N, VSL / TILE_M, BATCH), 256, 0, stream>>>(
        vis_pb, (long)VSL * HD, HD,
        Wlv2b, 0, HD,
        nullptr, 0, HD,
        bias_lv,
        fused_out, (long)ML * HD,
        vis, (long)VSL * HD,
        vis_out, (long)VSL * HD, HD);
}
